// MT5_6777458393615
// MI455X (gfx1250) — matrix operand order repaired and checked by reading, not compiled, not run on hardware
//
#include <hip/hip_runtime.h>
#include <hip/hip_bf16.h>

// ---------------------------------------------------------------------------
// Types for CDNA5 WMMA (wave32): v_wmma_f32_16x16x32_bf16
// ---------------------------------------------------------------------------
typedef __attribute__((ext_vector_type(16))) __bf16 v16bf;
typedef __attribute__((ext_vector_type(8)))  __bf16 v8bf;
typedef __attribute__((ext_vector_type(8)))  float  v8f;
typedef __attribute__((ext_vector_type(4)))  float  v4f;
typedef __attribute__((ext_vector_type(4)))  int    v4i;

union frag16 { v16bf v; v8bf h[2]; };

__device__ __forceinline__ v8f v8f_zero() {
  v8f z = {0.f,0.f,0.f,0.f,0.f,0.f,0.f,0.f};
  return z;
}

// ---------------------------------------------------------------------------
// CDNA5 async global->LDS path (ASYNCcnt-tracked DMA-style copies).
// Probe round 2 showed the builtin takes (v4i AS1*, v4i AS3*, imm, imm).
// Still guarded so the kernel builds on toolchains without the builtin.
// ---------------------------------------------------------------------------
#if defined(__has_builtin)
#  if __has_builtin(__builtin_amdgcn_global_load_async_to_lds_b128) && \
      __has_builtin(__builtin_amdgcn_s_wait_asynccnt)
#    define USE_ASYNC_LDS 1
#  endif
#endif
#ifndef USE_ASYNC_LDS
#  define USE_ASYNC_LDS 0
#endif

#if USE_ASYNC_LDS
typedef __attribute__((address_space(1))) v4i gv4i_t;
typedef __attribute__((address_space(3))) v4i lv4i_t;
__device__ __forceinline__ void async_copy_b128(const void* g, void* l) {
  __builtin_amdgcn_global_load_async_to_lds_b128(
      (gv4i_t*)(unsigned long long)g,
      (lv4i_t*)(unsigned int)(unsigned long long)l,
      0, 0);
}
__device__ __forceinline__ void async_wait0() {
  __builtin_amdgcn_s_wait_asynccnt(0);
}
#else
__device__ __forceinline__ void async_wait0() {}
#endif

// ---------------------------------------------------------------------------
// GEMM: C[M,N] = A[M,K] (f32 or bf16, row-major) @ W[K,N] (f32, row-major)
// Output TO = __bf16 (intermediates) or float (final).
// Block: 256 thr (8 waves). Block tile 128x128, wave tile 32x64, K-step 32.
// ---------------------------------------------------------------------------
template<typename TA, typename TO>
__global__ __launch_bounds__(256)
void gemm_wmma(const TA* __restrict__ A, const float* __restrict__ W,
               TO* __restrict__ C, int M, int N, int K) {
  __shared__ alignas(16) __bf16 sA [128][32];   // [m][k]
  __shared__ alignas(16) __bf16 sBt[128][32];   // [n][k]  (W transposed)

  const int tid  = threadIdx.x;
  const int bm   = blockIdx.x * 128;
  const int bn   = blockIdx.y * 128;
  const int wave = tid >> 5;
  const int lane = tid & 31;
  const int lh   = lane & 15;       // lane within half-wave
  const int hi   = lane >> 4;       // half-wave select
  const int wm   = (wave & 3) * 32; // wave M offset in block tile
  const int wn   = (wave >> 2) * 64;// wave N offset in block tile

  v8f acc[2][4];
#pragma unroll
  for (int i = 0; i < 2; ++i)
#pragma unroll
    for (int j = 0; j < 4; ++j) acc[i][j] = v8f_zero();

  const int arow = tid >> 1, acol = (tid & 1) * 16;   // A staging: 16 elems/thr
  const int wk   = tid >> 3, wn0  = (tid & 7) * 16;   // W staging: 16 elems/thr

  for (int kk = 0; kk < K; kk += 32) {
    // ---- stage A tile (128x32) as bf16 --------------------------------
    {
      const TA* src = A + (size_t)(bm + arow) * K + kk + acol;
      if constexpr (sizeof(TA) == 4) {
        __bf16 tmp[16];
        const v4f* p = (const v4f*)src;
#pragma unroll
        for (int v = 0; v < 4; ++v) {
          v4f f = p[v];
          tmp[v*4+0] = (__bf16)f.x; tmp[v*4+1] = (__bf16)f.y;
          tmp[v*4+2] = (__bf16)f.z; tmp[v*4+3] = (__bf16)f.w;
        }
        *(v8bf*)&sA[arow][acol]     = *(v8bf*)&tmp[0];
        *(v8bf*)&sA[arow][acol + 8] = *(v8bf*)&tmp[8];
      } else {
#if USE_ASYNC_LDS
        async_copy_b128(src,     &sA[arow][acol]);       // bf16 straight copy:
        async_copy_b128(src + 8, &sA[arow][acol + 8]);   // async DMA to LDS
#else
        const v8bf* p = (const v8bf*)src;
        *(v8bf*)&sA[arow][acol]     = p[0];
        *(v8bf*)&sA[arow][acol + 8] = p[1];
#endif
      }
    }
    // ---- stage W tile (32x128) transposed -> sBt[n][k] ----------------
    {
      const v4f* p = (const v4f*)(W + (size_t)(kk + wk) * N + bn + wn0);
#pragma unroll
      for (int v = 0; v < 4; ++v) {
        v4f f = p[v];
        sBt[wn0 + v*4 + 0][wk] = (__bf16)f.x;
        sBt[wn0 + v*4 + 1][wk] = (__bf16)f.y;
        sBt[wn0 + v*4 + 2][wk] = (__bf16)f.z;
        sBt[wn0 + v*4 + 3][wk] = (__bf16)f.w;
      }
    }
    if (kk + 32 < K) {  // prefetch next tiles (global_prefetch_b8)
      __builtin_prefetch(A + (size_t)(bm + arow) * K + kk + 32 + acol, 0, 1);
      __builtin_prefetch(W + (size_t)(kk + 32 + wk) * N + bn + wn0, 0, 1);
    }
    if constexpr (sizeof(TA) != 4) async_wait0();
    __syncthreads();

    // ---- fragments + 8 WMMAs -----------------------------------------
    frag16 afr[2], bfr[4];
#pragma unroll
    for (int mt = 0; mt < 2; ++mt) {        // A: lane=row, K split per ISA
      int row = wm + mt * 16 + lh;
      int k0  = hi * 8;
      afr[mt].h[0] = *(const v8bf*)&sA[row][k0];
      afr[mt].h[1] = *(const v8bf*)&sA[row][k0 + 16];
    }
#pragma unroll
    for (int nt = 0; nt < 4; ++nt) {        // B: lane=col, K split per ISA
      int n  = wn + nt * 16 + lh;
      int k0 = hi * 8;
      bfr[nt].h[0] = *(const v8bf*)&sBt[n][k0];
      bfr[nt].h[1] = *(const v8bf*)&sBt[n][k0 + 16];
    }
#pragma unroll
    for (int mt = 0; mt < 2; ++mt)
#pragma unroll
      for (int nt = 0; nt < 4; ++nt)
        acc[mt][nt] = __builtin_amdgcn_wmma_f32_16x16x32_bf16(
            false, afr[mt].v, false, bfr[nt].v, (short)0, acc[mt][nt],
            false, false);

    __syncthreads();
  }

  // ---- epilogue: C layout = lane:N, vgpr r:M (+8 for upper half) ------
#pragma unroll
  for (int mt = 0; mt < 2; ++mt) {
#pragma unroll
    for (int nt = 0; nt < 4; ++nt) {
      int col = bn + wn + nt * 16 + lh;
#pragma unroll
      for (int r = 0; r < 8; ++r) {
        int row = bm + wm + mt * 16 + hi * 8 + r;
        float vv = acc[mt][nt][r];
        if constexpr (sizeof(TO) == 4) C[(size_t)row * N + col] = vv;
        else                           C[(size_t)row * N + col] = (TO)vv;
      }
    }
  }
}

// ---------------------------------------------------------------------------
// Flash attention (T5: no 1/sqrt(dh), additive bias).
// Grid: (SQ/128, H, B). Block: 256 thr; wave w handles q rows [128*bx+16w, +16).
// Q/K/V are bf16 with layout [b*S + s][h*64 + d] (row stride 1024).
// ---------------------------------------------------------------------------
__global__ __launch_bounds__(256)
void attn_wmma(const __bf16* __restrict__ Q, const __bf16* __restrict__ Kw,
               const __bf16* __restrict__ Vw, const float* __restrict__ bias,
               __bf16* __restrict__ O) {
  constexpr int SQc = 1024, SKVc = 2048, DHc = 64;
  __shared__ alignas(16) __bf16 sK [64][64];     // [kv][dh] row-major
  __shared__ alignas(16) __bf16 sVt[64][64];     // [dh][kv] transposed
  __shared__ alignas(16) __bf16 sP [8][16][64];  // per-wave P relayout pad

  const int tid  = threadIdx.x;
  const int wave = tid >> 5, lane = tid & 31;
  const int lh = lane & 15, hi = lane >> 4;
  const int h = blockIdx.y, b = blockIdx.z;
  const int q0 = blockIdx.x * 128 + wave * 16;

  // ---- Q fragments (held for whole kv loop) ---------------------------
  frag16 qa[2];
  {
    const __bf16* qp = Q + (size_t)(b * SQc + q0 + lh) * 1024 + h * DHc;
#pragma unroll
    for (int ks = 0; ks < 2; ++ks) {
      int k0 = ks * 32 + hi * 8;
      qa[ks].h[0] = *(const v8bf*)(qp + k0);
      qa[ks].h[1] = *(const v8bf*)(qp + k0 + 16);
    }
  }

  float m[8], l[8];
  v8f acc[4];
#pragma unroll
  for (int r = 0; r < 8; ++r) { m[r] = -1e30f; l[r] = 0.f; }
#pragma unroll
  for (int dt = 0; dt < 4; ++dt) acc[dt] = v8f_zero();

  const int kvr = tid >> 2, d0 = (tid & 3) * 16;  // staging: 16 elems/thr

  for (int kv0 = 0; kv0 < SKVc; kv0 += 64) {
    __syncthreads();
    // ---- stage K (async DMA, row-major) and V (transposed) chunks ----
    {
      const __bf16* kp = Kw + (size_t)(b * SKVc + kv0 + kvr) * 1024 + h * DHc + d0;
#if USE_ASYNC_LDS
      async_copy_b128(kp,     &sK[kvr][d0]);
      async_copy_b128(kp + 8, &sK[kvr][d0 + 8]);
#else
      *(v8bf*)&sK[kvr][d0]     = *(const v8bf*)kp;
      *(v8bf*)&sK[kvr][d0 + 8] = *(const v8bf*)(kp + 8);
#endif
      const __bf16* vp = Vw + (size_t)(b * SKVc + kv0 + kvr) * 1024 + h * DHc + d0;
      v8bf v1 = *(const v8bf*)vp, v2 = *(const v8bf*)(vp + 8);
#pragma unroll
      for (int i = 0; i < 8; ++i) sVt[d0 + i][kvr] = v1[i];
#pragma unroll
      for (int i = 0; i < 8; ++i) sVt[d0 + 8 + i][kvr] = v2[i];
    }
    async_wait0();
    __syncthreads();

    // ---- S = Q @ K^T + bias (4 tiles of 16x16) -----------------------
    v8f s[4];
#pragma unroll
    for (int nt = 0; nt < 4; ++nt) {
      v8f st = v8f_zero();
#pragma unroll
      for (int ks = 0; ks < 2; ++ks) {
        frag16 kb;                      // B-frag: lane = kv column of K^T
        int n  = nt * 16 + lh;
        int k0 = ks * 32 + hi * 8;
        kb.h[0] = *(const v8bf*)&sK[n][k0];
        kb.h[1] = *(const v8bf*)&sK[n][k0 + 16];
        st = __builtin_amdgcn_wmma_f32_16x16x32_bf16(
            false, qa[ks].v, false, kb.v, (short)0, st, false, false);
      }
      const float* bp = bias + ((size_t)h * SQc + (q0 + hi * 8)) * SKVc
                             + kv0 + nt * 16 + lh;
#pragma unroll
      for (int r = 0; r < 8; ++r) st[r] += bp[(size_t)r * SKVc];
      s[nt] = st;
    }

    // ---- online softmax (per-row = per half-wave reductions) ---------
#pragma unroll
    for (int r = 0; r < 8; ++r) {
      float v = fmaxf(fmaxf(s[0][r], s[1][r]), fmaxf(s[2][r], s[3][r]));
#pragma unroll
      for (int off = 8; off >= 1; off >>= 1) v = fmaxf(v, __shfl_xor(v, off, 16));
      float mn = fmaxf(m[r], v);
      float alpha = __expf(m[r] - mn);
      m[r] = mn;
      float rsum = 0.f;
#pragma unroll
      for (int nt = 0; nt < 4; ++nt) {
        float p0 = __expf(s[nt][r] - mn);
        s[nt][r] = p0;
        rsum += p0;
      }
#pragma unroll
      for (int off = 8; off >= 1; off >>= 1) rsum += __shfl_xor(rsum, off, 16);
      l[r] = l[r] * alpha + rsum;
#pragma unroll
      for (int dt = 0; dt < 4; ++dt) acc[dt][r] *= alpha;
    }

    // ---- P (C-layout) -> LDS -> A-fragment layout --------------------
#pragma unroll
    for (int nt = 0; nt < 4; ++nt)
#pragma unroll
      for (int r = 0; r < 8; ++r)
        sP[wave][hi * 8 + r][nt * 16 + lh] = (__bf16)s[nt][r];
    // per-wave pad: same-wave LDS ops are in-order, no barrier needed

    // ---- O += P @ V ---------------------------------------------------
#pragma unroll
    for (int ks = 0; ks < 2; ++ks) {
      frag16 pa;
      int k0 = ks * 32 + hi * 8;
      pa.h[0] = *(const v8bf*)&sP[wave][lh][k0];
      pa.h[1] = *(const v8bf*)&sP[wave][lh][k0 + 16];
#pragma unroll
      for (int dt = 0; dt < 4; ++dt) {
        frag16 vb;                       // B-frag: lane = dh column
        int n   = dt * 16 + lh;
        int kb0 = ks * 32 + hi * 8;
        vb.h[0] = *(const v8bf*)&sVt[n][kb0];
        vb.h[1] = *(const v8bf*)&sVt[n][kb0 + 16];
        acc[dt] = __builtin_amdgcn_wmma_f32_16x16x32_bf16(
            false, pa.v, false, vb.v, (short)0, acc[dt], false, false);
      }
    }
  }

  // ---- epilogue: O[b*SQ+q][h*64+d] = acc / l --------------------------
#pragma unroll
  for (int dt = 0; dt < 4; ++dt) {
    int col = h * DHc + dt * 16 + lh;
#pragma unroll
    for (int r = 0; r < 8; ++r) {
      int row = b * SQc + q0 + hi * 8 + r;
      O[(size_t)row * 1024 + col] = (__bf16)(acc[dt][r] / l[r]);
    }
  }
}

// ---------------------------------------------------------------------------
// Launch: QKV projections -> flash attention -> output projection
// ---------------------------------------------------------------------------
extern "C" void kernel_launch(void* const* d_in, const int* in_sizes, int n_in,
                              void* d_out, int out_size, void* d_ws, size_t ws_size,
                              hipStream_t stream) {
  const float* X    = (const float*)d_in[0];  // [4,1024,1024]
  const float* E    = (const float*)d_in[1];  // [4,2048,1024]
  const float* bias = (const float*)d_in[2];  // [1,16,1024,2048]
  const float* Wq   = (const float*)d_in[3];
  const float* Wk   = (const float*)d_in[4];
  const float* Wv   = (const float*)d_in[5];
  const float* Wo   = (const float*)d_in[6];
  float* out = (float*)d_out;

  const size_t MQ = 4096, MKV = 8192, N = 1024, K = 1024;
  __bf16* Qws  = (__bf16*)d_ws;          // 4096*1024  (8 MB)
  __bf16* Kws  = Qws + MQ * N;           // 8192*1024  (16 MB)
  __bf16* Vws  = Kws + MKV * N;          // 8192*1024  (16 MB)
  __bf16* AOws = Vws + MKV * N;          // 4096*1024  (8 MB)

  dim3 blk(256);
  gemm_wmma<float, __bf16><<<dim3(MQ / 128,  N / 128), blk, 0, stream>>>(X, Wq, Qws,  (int)MQ,  (int)N, (int)K);
  gemm_wmma<float, __bf16><<<dim3(MKV / 128, N / 128), blk, 0, stream>>>(E, Wk, Kws,  (int)MKV, (int)N, (int)K);
  gemm_wmma<float, __bf16><<<dim3(MKV / 128, N / 128), blk, 0, stream>>>(E, Wv, Vws,  (int)MKV, (int)N, (int)K);
  attn_wmma<<<dim3(1024 / 128, 16, 4), blk, 0, stream>>>(Qws, Kws, Vws, bias, AOws);
  gemm_wmma<__bf16, float><<<dim3(MQ / 128,  N / 128), blk, 0, stream>>>(AOws, Wo, out, (int)MQ, (int)N, (int)K);
}
